// UNet3DModel_83923660964189
// MI455X (gfx1250) — matrix operand order repaired and checked by reading, not compiled, not run on hardware
//
#include <hip/hip_runtime.h>
#include <hip/hip_bf16.h>

// ---------------------------------------------------------------------------
// Types for CDNA5 WMMA (gfx1250): V_WMMA_F32_16X16X32_BF16
// ---------------------------------------------------------------------------
typedef __attribute__((ext_vector_type(16))) __bf16 v16bf;
typedef __attribute__((ext_vector_type(8)))  float  v8f;

union FragBF {
    uint4 q[2];   // 2 x 16B = 32B = 16 bf16
    v16bf v;
};

__device__ __forceinline__ __bf16 f2bf(float f) {
    unsigned int x = __float_as_uint(f);
    x += 0x7fffu + ((x >> 16) & 1u);          // round-to-nearest-even
    unsigned short h = (unsigned short)(x >> 16);
    __bf16 r;
    __builtin_memcpy(&r, &h, 2);
    return r;
}

__device__ __forceinline__ float silu(float v) {
    return v * (1.0f / (1.0f + __expf(-v)));
}

// ---------------------------------------------------------------------------
// WMMA GEMM:  C[M,Nc] = scale * (A[M,K](bf16) * Bt[Nc,K]^T(bf16)) + bias[n] (+ add)
//   Block: 256 threads = 8 waves; block tile 128(M) x 128(N); wave tile 32x64.
//   Per K-step each wave issues 8 v_wmma_f32_16x16x32_bf16, reusing both A
//   fragments across 4 B fragments (halves A HBM re-reads vs a 32x32 tile).
//   A lane layout (16-bit A 16x32): lane<16 -> M=lane, K={0..7,16..23};
//                                   lane>=16 -> M=lane-16, K={8..15,24..31}.
//   B lane layout (16-bit B 32x16): lane<16 -> N=lane, K={0..7,16..23};
//                                   lane>=16 -> N=lane-16, K={8..15,24..31}.
//   C/D layout: VGPR j: lanes0-15 (M=j,N=lane), lanes16-31 (M=j+8,N=lane-16).
// ---------------------------------------------------------------------------
__global__ __launch_bounds__(256)
void k_wmma_gemm(const __bf16* __restrict__ A, int lda,
                 const __bf16* __restrict__ Bt, int ldb,
                 const float* __restrict__ bias, float scale,
                 const float* __restrict__ addend,   // nullable, [M,Nc]
                 float* __restrict__ C,
                 int M, int Nc, int K)
{
    const int lane = threadIdx.x & 31;
    const int wid  = threadIdx.x >> 5;
    const int l15  = lane & 15;
    const int half = lane >> 4;

    const int m0 = blockIdx.x * 128 + (wid & 3) * 32;
    const int n0 = blockIdx.y * 128 + (wid >> 2) * 64;

    v8f acc[2][4];
    #pragma unroll
    for (int i = 0; i < 2; ++i)
        #pragma unroll
        for (int j = 0; j < 4; ++j) acc[i][j] = {};

    const __bf16* ap0 = A + (size_t)(m0 + l15) * lda;
    const __bf16* ap1 = ap0 + (size_t)16 * lda;
    const __bf16* bp[4];
    #pragma unroll
    for (int j = 0; j < 4; ++j) bp[j] = Bt + (size_t)(n0 + j * 16 + l15) * ldb;

    const int aoff = half * 8;    // A: K sub-block within 32
    const int boff = half * 8;    // B: K sub-block within 32 (same as A)

    for (int kk = 0; kk < K; kk += 32) {
        FragBF a[2], b[4];
        a[0].q[0] = *(const uint4*)(ap0 + kk + aoff);
        a[0].q[1] = *(const uint4*)(ap0 + kk + aoff + 16);
        a[1].q[0] = *(const uint4*)(ap1 + kk + aoff);
        a[1].q[1] = *(const uint4*)(ap1 + kk + aoff + 16);
        #pragma unroll
        for (int j = 0; j < 4; ++j) {
            b[j].q[0] = *(const uint4*)(bp[j] + kk + boff);
            b[j].q[1] = *(const uint4*)(bp[j] + kk + boff + 16);
        }
        // prefetch next K-blocks of the streamed A operand into near cache
        __builtin_prefetch(ap0 + kk + 128, 0, 3);
        __builtin_prefetch(ap1 + kk + 128, 0, 3);

        #pragma unroll
        for (int i = 0; i < 2; ++i) {
            #pragma unroll
            for (int j = 0; j < 4; ++j) {
                acc[i][j] = __builtin_amdgcn_wmma_f32_16x16x32_bf16(
                    false, a[i].v, false, b[j].v, (short)0, acc[i][j], false, false);
            }
        }
    }

    #pragma unroll
    for (int i = 0; i < 2; ++i) {
        #pragma unroll
        for (int j = 0; j < 4; ++j) {
            const int mb  = m0 + i * 16;
            const int col = n0 + j * 16 + l15;
            const float bb = bias ? bias[col] : 0.0f;
            #pragma unroll
            for (int r = 0; r < 8; ++r) {
                const int row = mb + r + (half ? 8 : 0);
                const size_t o = (size_t)row * Nc + col;
                float v = acc[i][j][r] * scale + bb;
                if (addend) v += addend[o];
                C[o] = v;
            }
        }
    }
}

// ---------------------------------------------------------------------------
// GroupNorm(32 groups) [+ FiLM] + SiLU.  One wave per node, one lane per group
// (wave32-native: 32 lanes == 32 groups, group channels are contiguous).
// film (nullable): [2*C] = {scale[C], shift[C]}; y = GN(x)*(1+scale)+shift then SiLU.
// ---------------------------------------------------------------------------
__global__ void k_gn(const float* __restrict__ x, const float* __restrict__ g,
                     const float* __restrict__ b, const float* __restrict__ film,
                     float* __restrict__ y, int N, int C)
{
    const int n = blockIdx.x * 8 + (threadIdx.x >> 5);
    if (n >= N) return;
    const int lane = threadIdx.x & 31;
    const int cg = C >> 5;
    const float* xr = x + (size_t)n * C + lane * cg;
    float s = 0.0f, s2 = 0.0f;
    for (int i = 0; i < cg; ++i) { float v = xr[i]; s += v; s2 += v * v; }
    const float mu  = s / (float)cg;
    const float var = s2 / (float)cg - mu * mu;
    const float inv = rsqrtf(var + 1e-5f);
    float* yr = y + (size_t)n * C + lane * cg;
    for (int i = 0; i < cg; ++i) {
        const int c = lane * cg + i;
        float gg = g[c], bb = b[c];
        if (film) {
            const float sc = 1.0f + film[c];
            gg *= sc;
            bb = bb * sc + film[C + c];
        }
        const float v = (xr[i] - mu) * inv * gg + bb;
        yr[i] = silu(v);
    }
}

// ---------------------------------------------------------------------------
// Timestep embedding + time MLP (single block).  s_emb = silu(MLP(t))  [256]
// ---------------------------------------------------------------------------
__global__ void k_time(const float* __restrict__ t_in,
                       const float* __restrict__ w1, const float* __restrict__ b1,
                       const float* __restrict__ w2, const float* __restrict__ b2,
                       float* __restrict__ s_emb)
{
    __shared__ float e0[128];
    __shared__ float e1[256];
    const float t = t_in[0];
    const int tid = threadIdx.x;
    if (tid < 128) {
        const int i = tid & 63;
        const float f = __expf(-logf(10000.0f) * (float)i / 64.0f);
        const float a = t * f;
        e0[tid] = (tid < 64) ? __cosf(a) : __sinf(a);
    }
    __syncthreads();
    float acc = b1[tid];
    for (int k = 0; k < 128; ++k) acc += e0[k] * w1[k * 256 + tid];
    e1[tid] = silu(acc);
    __syncthreads();
    float a2 = b2[tid];
    for (int k = 0; k < 256; ++k) a2 += e1[k] * w2[k * 256 + tid];
    s_emb[tid] = silu(a2);
}

// eo[j] = s_emb . embW[:,j] + embB[j],  j < 2*Co
__global__ void k_emb_out(const float* __restrict__ s, const float* __restrict__ w,
                          const float* __restrict__ b, float* __restrict__ eo, int n2co)
{
    const int j = blockIdx.x * 256 + threadIdx.x;
    if (j >= n2co) return;
    float acc = b[j];
    for (int k = 0; k < 256; ++k) acc += s[k] * w[(size_t)k * n2co + j];
    eo[j] = acc;
}

// ---------------------------------------------------------------------------
// Scatter gathered neighbor features into (node, edge_type) bins (f32 atomics)
// ---------------------------------------------------------------------------
__global__ void k_scatter4(const float* __restrict__ x, const int* __restrict__ ei,
                           const int* __restrict__ et, float* __restrict__ binned,
                           long long E, int C, int ldk)
{
    const int C4 = C >> 2;
    const long long total = E * (long long)C4;
    const long long stride = (long long)gridDim.x * blockDim.x;
    for (long long i = blockIdx.x * (long long)blockDim.x + threadIdx.x; i < total; i += stride) {
        const long long e = i / C4;
        const int c = (int)(i - e * C4) * 4;
        const int row = ei[e], col = ei[E + e], t = et[e];
        const float4 v = *(const float4*)(x + (size_t)col * C + c);
        float* dst = binned + (size_t)row * ldk + t * C + c;
        atomicAdd(dst + 0, v.x);
        atomicAdd(dst + 1, v.y);
        atomicAdd(dst + 2, v.z);
        atomicAdd(dst + 3, v.w);
    }
}

__global__ void k_scatter1(const float* __restrict__ x, const int* __restrict__ ei,
                           const int* __restrict__ et, float* __restrict__ binned,
                           long long E, int C, int ldk)
{
    const long long stride = (long long)gridDim.x * blockDim.x;
    for (long long e = blockIdx.x * (long long)blockDim.x + threadIdx.x; e < E; e += stride) {
        const int row = ei[e], col = ei[E + e], t = et[e];
        for (int c = 0; c < C; ++c)
            atomicAdd(binned + (size_t)row * ldk + t * C + c, x[(size_t)col * C + c]);
    }
}

// ---------------------------------------------------------------------------
// Conversions / transposes
// ---------------------------------------------------------------------------
__global__ void k_cvt_bf16(const float* __restrict__ src, __bf16* __restrict__ dst, long long n)
{
    const long long stride = (long long)gridDim.x * blockDim.x;
    for (long long i = blockIdx.x * (long long)blockDim.x + threadIdx.x; i < n; i += stride)
        dst[i] = f2bf(src[i]);
}

// w [K,Co] f32 (row-major) -> wt [Co,Kp] bf16 (row-major, zero-padded K)
__global__ void k_wt_t(const float* __restrict__ w, __bf16* __restrict__ wt,
                       int K, int Co, int Kp)
{
    const long long total = (long long)Co * Kp;
    const long long stride = (long long)gridDim.x * blockDim.x;
    for (long long i = blockIdx.x * (long long)blockDim.x + threadIdx.x; i < total; i += stride) {
        const int co = (int)(i / Kp);
        const int k  = (int)(i - (long long)co * Kp);
        wt[i] = (k < K) ? f2bf(w[(size_t)k * Co + co]) : f2bf(0.0f);
    }
}

// ---------------------------------------------------------------------------
// Pooling / gather / concat helpers
// ---------------------------------------------------------------------------
__global__ void k_count(const int* __restrict__ c2p, float* __restrict__ cnt, int N)
{
    const int i = blockIdx.x * 256 + threadIdx.x;
    if (i < N) atomicAdd(&cnt[c2p[i]], 1.0f);
}

__global__ void k_pool_scatter(const float* __restrict__ x, const int* __restrict__ c2p,
                               float* __restrict__ sum, long long NC, int C)
{
    const long long stride = (long long)gridDim.x * blockDim.x;
    for (long long i = blockIdx.x * (long long)blockDim.x + threadIdx.x; i < NC; i += stride) {
        const long long n = i / C;
        const int c = (int)(i - n * C);
        atomicAdd(&sum[(size_t)c2p[n] * C + c], x[i]);
    }
}

__global__ void k_pool_div(float* __restrict__ sum, const float* __restrict__ cnt,
                           long long NpC, int C)
{
    const long long stride = (long long)gridDim.x * blockDim.x;
    for (long long i = blockIdx.x * (long long)blockDim.x + threadIdx.x; i < NpC; i += stride)
        sum[i] /= fmaxf(cnt[i / C], 1.0f);
}

__global__ void k_gather(const float* __restrict__ x, const int* __restrict__ idx,
                         float* __restrict__ y, long long total, int C)
{
    const long long stride = (long long)gridDim.x * blockDim.x;
    for (long long i = blockIdx.x * (long long)blockDim.x + threadIdx.x; i < total; i += stride) {
        const long long n = i / C;
        const int c = (int)(i - n * C);
        y[i] = x[(size_t)idx[n] * C + c];
    }
}

__global__ void k_concat(const float* __restrict__ a, int Ca,
                         const float* __restrict__ b, int Cb,
                         float* __restrict__ y, long long total)
{
    const int Ct = Ca + Cb;
    const long long stride = (long long)gridDim.x * blockDim.x;
    for (long long i = blockIdx.x * (long long)blockDim.x + threadIdx.x; i < total; i += stride) {
        const long long n = i / Ct;
        const int c = (int)(i - n * Ct);
        y[i] = (c < Ca) ? a[n * (long long)Ca + c] : b[n * (long long)Cb + (c - Ca)];
    }
}

// Final out_conv: [N,896] f32 x [896,3] -> d_out [N,3]  (Co=3: scalar GEMV)
__global__ void k_outconv(const float* __restrict__ binned, const float* __restrict__ w,
                          const float* __restrict__ b, float* __restrict__ out,
                          int N, int K, float scale)
{
    const int n = blockIdx.x * 256 + threadIdx.x;
    if (n >= N) return;
    const float* r = binned + (size_t)n * K;
    float a0 = 0.f, a1 = 0.f, a2 = 0.f;
    for (int k = 0; k < K; ++k) {
        const float v = r[k];
        a0 += v * w[k * 3 + 0];
        a1 += v * w[k * 3 + 1];
        a2 += v * w[k * 3 + 2];
    }
    out[n * 3 + 0] = a0 * scale + b[0];
    out[n * 3 + 1] = a1 * scale + b[1];
    out[n * 3 + 2] = a2 * scale + b[2];
}

// ===========================================================================
// Host orchestration
// ===========================================================================
extern "C" void kernel_launch(void* const* d_in, const int* in_sizes, int n_in,
                              void* d_out, int out_size, void* d_ws, size_t ws_size,
                              hipStream_t stream)
{
    const float* x_in   = (const float*)d_in[0];
    const float* tsteps = (const float*)d_in[1];
    const int* ei6 = (const int*)d_in[2];
    const int* et6 = (const int*)d_in[3];
    const int* ei5 = (const int*)d_in[4];
    const int* et5 = (const int*)d_in[5];
    const int* ei4 = (const int*)d_in[6];
    const int* et4 = (const int*)d_in[7];
    const int* c2p6 = (const int*)d_in[8];
    const int* c2p5 = (const int*)d_in[9];

    const long long E6 = in_sizes[3], E5 = in_sizes[5], E4 = in_sizes[7];
    const int N6 = in_sizes[0] / 3;
    const int N5 = in_sizes[9];
    const int N4 = (int)(E4 / 7);

    // ---- walk parameter leaves in setup_inputs() insertion order -----------
    int cur = 10;
    auto NXT = [&]() { return (const float*)d_in[cur++]; };
    struct Lin { const float* w; const float* b; };
    auto getLin = [&]() { Lin l; l.w = NXT(); l.b = NXT(); return l; };
    struct Res {
        const float *n1g, *n1b, *n2g, *n2b;
        Lin conv1, emb, conv2, skip;
        bool has_skip; int ci, co;
    };
    auto getRes = [&](int ci, int co) {
        Res r; r.ci = ci; r.co = co;
        r.n1g = NXT(); r.n1b = NXT();
        r.conv1 = getLin(); r.emb = getLin();
        r.n2g = NXT(); r.n2b = NXT();
        r.conv2 = getLin();
        r.has_skip = (ci != co);
        if (r.has_skip) r.skip = getLin();
        return r;
    };
    Lin time1 = getLin(), time2 = getLin(), input_conv = getLin();
    Res d6a = getRes(128, 128), d6b = getRes(128, 128);
    Res d5a = getRes(128, 256), d5b = getRes(256, 256);
    Res d4a = getRes(256, 512), d4b = getRes(512, 512);
    Lin ds6 = getLin(), ds5 = getLin();
    Res mid1 = getRes(512, 512), mid2 = getRes(512, 512);
    Res u4a = getRes(1024, 512), u4b = getRes(1024, 512), u4c = getRes(768, 512);
    Res u5a = getRes(768, 256),  u5b = getRes(512, 256),  u5c = getRes(384, 256);
    Res u6a = getRes(384, 128),  u6b = getRes(256, 128),  u6c = getRes(256, 128);
    Lin us4 = getLin(), us5 = getLin();
    const float* outg = NXT();
    const float* outb = NXT();
    Lin out_conv = getLin();

    // ---- workspace bump allocator ------------------------------------------
    char* wsb = (char*)d_ws;
    size_t off = 0;
    auto alloc = [&](size_t bytes) { void* p = wsb + off; off += (bytes + 255) & ~(size_t)255; return p; };

    float* s_emb = (float*)alloc(256 * 4);
    float* eo    = (float*)alloc(1024 * 4);
    float* hs0 = (float*)alloc((size_t)N6 * 128 * 4);
    float* hs1 = (float*)alloc((size_t)N6 * 128 * 4);
    float* hs2 = (float*)alloc((size_t)N6 * 128 * 4);
    float* hs3 = (float*)alloc((size_t)N5 * 128 * 4);
    float* hs4 = (float*)alloc((size_t)N5 * 256 * 4);
    float* hs5 = (float*)alloc((size_t)N5 * 256 * 4);
    float* hs6 = (float*)alloc((size_t)N4 * 256 * 4);
    float* hs7 = (float*)alloc((size_t)N4 * 512 * 4);
    float* hs8 = (float*)alloc((size_t)N4 * 512 * 4);
    const size_t SA = (size_t)N6 * 384;              // widest activation (concat at depth 6)
    float* bufA   = (float*)alloc(SA * 4);
    float* bufB   = (float*)alloc(SA * 4);
    float* bufN   = (float*)alloc(SA * 4);
    float* bufH1  = (float*)alloc((size_t)N6 * 128 * 4);
    float* bufSkip= (float*)alloc((size_t)N6 * 128 * 4);
    size_t poolsz = (size_t)N5 * 128; { size_t p2 = (size_t)N4 * 256; if (p2 > poolsz) poolsz = p2; }
    float* pooled = (float*)alloc(poolsz * 4);
    float* cnt    = (float*)alloc((size_t)N5 * 4);
    size_t BK = (size_t)N6 * 2688;                   // largest binned: depth6, C=384
    { size_t b2 = (size_t)N5 * 5376, b3 = (size_t)N4 * 7168;
      if (b2 > BK) BK = b2; if (b3 > BK) BK = b3; }
    float*  binned   = (float*)alloc(BK * 4);
    __bf16* binnedBf = (__bf16*)alloc(BK * 2);
    __bf16* xBf      = (__bf16*)alloc(SA * 2);
    __bf16* wtBf     = (__bf16*)alloc((size_t)7168 * 512 * 2);
    (void)ws_size; (void)n_in; (void)out_size;

    const float invs7 = 0.37796447300922720f;   // 1/sqrt(7)
    auto nblocks = [](long long n) { return (unsigned)((n + 255) / 256); };

    // ---- composite ops ------------------------------------------------------
    auto graph_conv = [&](const float* xin, int N, int C, int Co, const Lin& L,
                          const int* ei, const int* et, long long E,
                          float* out, const float* add) {
        const int K  = 7 * C;
        const int Kp = (K + 31) & ~31;
        hipMemsetAsync(binned, 0, (size_t)N * Kp * sizeof(float), stream);
        if ((C & 3) == 0) {
            k_scatter4<<<nblocks(E * (C >> 2)), 256, 0, stream>>>(xin, ei, et, binned, E, C, Kp);
        } else {
            k_scatter1<<<nblocks(E), 256, 0, stream>>>(xin, ei, et, binned, E, C, Kp);
        }
        const long long nb = (long long)N * Kp;
        k_cvt_bf16<<<nblocks(nb), 256, 0, stream>>>(binned, binnedBf, nb);
        k_wt_t<<<nblocks((long long)Co * Kp), 256, 0, stream>>>(L.w, wtBf, K, Co, Kp);
        dim3 g((unsigned)(N / 128), (unsigned)(Co / 128));
        k_wmma_gemm<<<g, 256, 0, stream>>>(binnedBf, Kp, wtBf, Kp, L.b, invs7, add, out, N, Co, Kp);
    };

    auto lin_gemm = [&](const float* xin, int N, int Ci, int Co, const Lin& L, float* out) {
        const long long n = (long long)N * Ci;
        k_cvt_bf16<<<nblocks(n), 256, 0, stream>>>(xin, xBf, n);
        k_wt_t<<<nblocks((long long)Co * Ci), 256, 0, stream>>>(L.w, wtBf, Ci, Co, Ci);
        dim3 g((unsigned)(N / 128), (unsigned)(Co / 128));
        k_wmma_gemm<<<g, 256, 0, stream>>>(xBf, Ci, wtBf, Ci, L.b, 1.0f, nullptr, out, N, Co, Ci);
    };

    auto res_block = [&](const float* xin, int N, const Res& r,
                         const int* ei, const int* et, long long E, float* out) {
        k_emb_out<<<nblocks(2 * r.co), 256, 0, stream>>>(s_emb, r.emb.w, r.emb.b, eo, 2 * r.co);
        k_gn<<<(N + 7) / 8, 256, 0, stream>>>(xin, r.n1g, r.n1b, nullptr, bufN, N, r.ci);
        graph_conv(bufN, N, r.ci, r.co, r.conv1, ei, et, E, bufH1, nullptr);
        k_gn<<<(N + 7) / 8, 256, 0, stream>>>(bufH1, r.n2g, r.n2b, eo, bufN, N, r.co);
        const float* sk = xin;
        if (r.has_skip) { lin_gemm(xin, N, r.ci, r.co, r.skip, bufSkip); sk = bufSkip; }
        graph_conv(bufN, N, r.co, r.co, r.conv2, ei, et, E, out, sk);
    };

    auto pool = [&](const float* hc, int Nc_, int C, const int* c2p, int Np) {
        hipMemsetAsync(pooled, 0, (size_t)Np * C * 4, stream);
        hipMemsetAsync(cnt, 0, (size_t)Np * 4, stream);
        k_count<<<nblocks(Nc_), 256, 0, stream>>>(c2p, cnt, Nc_);
        k_pool_scatter<<<nblocks((long long)Nc_ * C), 256, 0, stream>>>(hc, c2p, pooled, (long long)Nc_ * C, C);
        k_pool_div<<<nblocks((long long)Np * C), 256, 0, stream>>>(pooled, cnt, (long long)Np * C, C);
    };

    auto cat = [&](const float* a, int Ca, const float* b, int Cb, int N, float* y) {
        const long long n = (long long)N * (Ca + Cb);
        k_concat<<<nblocks(n), 256, 0, stream>>>(a, Ca, b, Cb, y, n);
    };

    // ---- forward pass -------------------------------------------------------
    k_time<<<1, 256, 0, stream>>>(tsteps, time1.w, time1.b, time2.w, time2.b, s_emb);

    graph_conv(x_in, N6, 3, 128, input_conv, ei6, et6, E6, hs0, nullptr);
    res_block(hs0, N6, d6a, ei6, et6, E6, hs1);
    res_block(hs1, N6, d6b, ei6, et6, E6, hs2);
    pool(hs2, N6, 128, c2p6, N5);
    graph_conv(pooled, N5, 128, 128, ds6, ei5, et5, E5, hs3, nullptr);
    res_block(hs3, N5, d5a, ei5, et5, E5, hs4);
    res_block(hs4, N5, d5b, ei5, et5, E5, hs5);
    pool(hs5, N5, 256, c2p5, N4);
    graph_conv(pooled, N4, 256, 256, ds5, ei4, et4, E4, hs6, nullptr);
    res_block(hs6, N4, d4a, ei4, et4, E4, hs7);
    res_block(hs7, N4, d4b, ei4, et4, E4, hs8);

    res_block(hs8, N4, mid1, ei4, et4, E4, bufA);
    res_block(bufA, N4, mid2, ei4, et4, E4, bufB);

    cat(bufB, 512, hs8, 512, N4, bufA);  res_block(bufA, N4, u4a, ei4, et4, E4, bufB);
    cat(bufB, 512, hs7, 512, N4, bufA);  res_block(bufA, N4, u4b, ei4, et4, E4, bufB);
    cat(bufB, 512, hs6, 256, N4, bufA);  res_block(bufA, N4, u4c, ei4, et4, E4, bufB);

    k_gather<<<nblocks((long long)N5 * 512), 256, 0, stream>>>(bufB, c2p5, bufA, (long long)N5 * 512, 512);
    graph_conv(bufA, N5, 512, 512, us4, ei5, et5, E5, bufB, nullptr);

    cat(bufB, 512, hs5, 256, N5, bufA);  res_block(bufA, N5, u5a, ei5, et5, E5, bufB);
    cat(bufB, 256, hs4, 256, N5, bufA);  res_block(bufA, N5, u5b, ei5, et5, E5, bufB);
    cat(bufB, 256, hs3, 128, N5, bufA);  res_block(bufA, N5, u5c, ei5, et5, E5, bufB);

    k_gather<<<nblocks((long long)N6 * 256), 256, 0, stream>>>(bufB, c2p6, bufA, (long long)N6 * 256, 256);
    graph_conv(bufA, N6, 256, 256, us5, ei6, et6, E6, bufB, nullptr);

    cat(bufB, 256, hs2, 128, N6, bufA);  res_block(bufA, N6, u6a, ei6, et6, E6, bufB);
    cat(bufB, 128, hs1, 128, N6, bufA);  res_block(bufA, N6, u6b, ei6, et6, E6, bufB);
    cat(bufB, 128, hs0, 128, N6, bufA);  res_block(bufA, N6, u6c, ei6, et6, E6, bufB);

    // out head: silu(GN(h)) -> graph_conv to 3 channels (scalar GEMV, Co=3)
    k_gn<<<(N6 + 7) / 8, 256, 0, stream>>>(bufB, outg, outb, nullptr, bufA, N6, 128);
    hipMemsetAsync(binned, 0, (size_t)N6 * 896 * sizeof(float), stream);
    k_scatter4<<<nblocks(E6 * 32), 256, 0, stream>>>(bufA, ei6, et6, binned, E6, 128, 896);
    k_outconv<<<nblocks(N6), 256, 0, stream>>>(binned, out_conv.w, out_conv.b, (float*)d_out, N6, 896, invs7);
}
